// MultiHeadSelfAttention_37400575213821
// MI455X (gfx1250) — compile-verified, matrix operand order checked by reading, not run on hardware
//
#include <hip/hip_runtime.h>
#include <hip/hip_bf16.h>

// ---------------------------------------------------------------------------
// MHSA for MI455X (gfx1250): bf16 WMMA GEMMs + flash-attention.
// B=2, L=2048, H=1024, NH=16, HS=64, ALIBI_HEADS=8.
// Round 3: fix async-to-LDS builtin arg types (v4i addrspace(1)/(3) pointers).
// ---------------------------------------------------------------------------

typedef __bf16 bf16_t;
typedef __attribute__((ext_vector_type(16))) __bf16 v16bf;
typedef __attribute__((ext_vector_type(8)))  float  v8f;
typedef __attribute__((ext_vector_type(4)))  int    v4i_t;

#define C_B   2
#define C_L   2048
#define C_H   1024
#define C_NH  16
#define C_HS  64

#if __has_builtin(__builtin_amdgcn_global_load_async_to_lds_b128)
#define HAVE_ASYNC_LDS 1
#endif

typedef __attribute__((address_space(1))) v4i_t g_v4i;   // global int4
typedef __attribute__((address_space(3))) v4i_t l_v4i;   // LDS int4

// 16-byte global -> LDS copy: async DMA when available, sync fallback.
static __device__ __forceinline__ void cp16(const bf16_t* g, bf16_t* l) {
#ifdef HAVE_ASYNC_LDS
  __builtin_amdgcn_global_load_async_to_lds_b128(
      (g_v4i*)(void*)g, (l_v4i*)(void*)l, 0, 0);
#else
  *(uint4*)l = *(const uint4*)g;
#endif
}

static __device__ __forceinline__ void wait_async_all() {
#ifdef HAVE_ASYNC_LDS
#if __has_builtin(__builtin_amdgcn_s_wait_asynccnt)
  __builtin_amdgcn_s_wait_asynccnt(0);
#else
  asm volatile("s_wait_asynccnt 0" ::: "memory");
#endif
#endif
}

// Load one 16x32 bf16 A/B fragment half-pair: lane supplies two contiguous
// 16-byte groups (K = kb..kb+7 and K = kb+16..kb+23 for this lane).
static __device__ __forceinline__ v16bf ld_frag(const bf16_t* p0, const bf16_t* p1) {
  union { uint4 u[2]; v16bf v; } f;
  f.u[0] = *(const uint4*)p0;
  f.u[1] = *(const uint4*)p1;
  return f.v;
}

static __device__ __forceinline__ v8f wmma_bf16(v16bf a, v16bf b, v8f c) {
  return __builtin_amdgcn_wmma_f32_16x16x32_bf16(false, a, false, b, (short)0, c, false, false);
}

// ---------------------------------------------------------------------------
// Conversion / transpose helpers
// ---------------------------------------------------------------------------
__global__ void k_cvt_bf16(const float* __restrict__ in, bf16_t* __restrict__ out, int n) {
  int i = blockIdx.x * 256 + threadIdx.x;
  if (i < n) out[i] = (bf16_t)in[i];
}

// in [R][C] f32 -> out [C][R] bf16
__global__ void k_transpose_bf16(const float* __restrict__ in, bf16_t* __restrict__ out,
                                 int R, int C) {
  int i = blockIdx.x * 256 + threadIdx.x;
  if (i >= R * C) return;
  int r = i / C, c = i - r * C;
  out[(size_t)c * R + r] = (bf16_t)in[(size_t)r * C + c];
}

// ---------------------------------------------------------------------------
// Tiled bf16 GEMM:  C[M,N] = A[M,K] * Bt[N,K]^T + bias[N]
//   mode 0: scatter result into per-head Q[bh,l,hs], K[bh,l,hs], Vt[bh,hs,l] (bf16)
//   mode 1: store fp32 row-major to Cout
// Block: 256 threads (8 waves), tile 128x128, K-step 64. Wave = 32x64 (2x4 subtiles).
// Double-buffered LDS with async DMA prefetch of the next K-tile.
// ---------------------------------------------------------------------------
#define TM 128
#define TN 128
#define TK 64

__global__ __launch_bounds__(256) void k_gemm_bf16(
    const bf16_t* __restrict__ A, const bf16_t* __restrict__ Bt,
    const float* __restrict__ bias, int M, int N, int K, int mode,
    bf16_t* __restrict__ Qb, bf16_t* __restrict__ Kb, bf16_t* __restrict__ Vt,
    float* __restrict__ Cout) {
  __shared__ __align__(16) bf16_t Asm[2][TM][TK];
  __shared__ __align__(16) bf16_t Bsm[2][TN][TK];

  const int tid  = threadIdx.x;
  const int lane = tid & 31;
  const int wave = tid >> 5;        // 0..7
  const int wm   = wave & 3;        // 4 waves over M
  const int wn   = wave >> 2;       // 2 waves over N
  const int m0   = blockIdx.y * TM;
  const int n0   = blockIdx.x * TN;
  const int lm   = lane & 15;
  const int hi   = lane >> 4;
  const int kh   = hi * 8;

  auto load_tile = [&](int k0, int buf) {
#pragma unroll
    for (int i = 0; i < 4; ++i) {           // 1024 16B chunks / 256 threads
      int c = tid + 256 * i;
      int r = c >> 3, cc = (c & 7) * 8;
      cp16(&A[(size_t)(m0 + r) * K + k0 + cc], &Asm[buf][r][cc]);
      cp16(&Bt[(size_t)(n0 + r) * K + k0 + cc], &Bsm[buf][r][cc]);
    }
  };

  v8f acc[2][4];
#pragma unroll
  for (int i = 0; i < 2; ++i)
#pragma unroll
    for (int j = 0; j < 4; ++j) acc[i][j] = (v8f)0.0f;

  load_tile(0, 0);
  wait_async_all();
  __syncthreads();

  int cur = 0;
  for (int k0 = 0; k0 < K; k0 += TK) {
    if (k0 + TK < K) load_tile(k0 + TK, cur ^ 1);   // DMA overlaps compute
#pragma unroll
    for (int ks = 0; ks < 2; ++ks) {
      v16bf af[2], bfv[4];
#pragma unroll
      for (int i = 0; i < 2; ++i) {
        const bf16_t* p = &Asm[cur][wm * 32 + i * 16 + lm][ks * 32 + kh];
        af[i] = ld_frag(p, p + 16);
      }
#pragma unroll
      for (int j = 0; j < 4; ++j) {
        const bf16_t* p = &Bsm[cur][wn * 64 + j * 16 + lm][ks * 32 + kh];
        bfv[j] = ld_frag(p, p + 16);
      }
#pragma unroll
      for (int i = 0; i < 2; ++i)
#pragma unroll
        for (int j = 0; j < 4; ++j) acc[i][j] = wmma_bf16(af[i], bfv[j], acc[i][j]);
    }
    wait_async_all();
    __syncthreads();
    cur ^= 1;
  }

  // epilogue
#pragma unroll
  for (int i = 0; i < 2; ++i)
#pragma unroll
    for (int j = 0; j < 4; ++j)
#pragma unroll
      for (int r = 0; r < 8; ++r) {
        int m = m0 + wm * 32 + i * 16 + r + 8 * hi;
        int n = n0 + wn * 64 + j * 16 + lm;
        float v = acc[i][j][r] + bias[n];
        if (mode == 1) {
          Cout[(size_t)m * N + n] = v;
        } else {
          int head  = n / (3 * C_HS);
          int rr    = n - head * (3 * C_HS);
          int which = rr / C_HS;
          int d     = rr - which * C_HS;
          int b     = m >> 11;          // /2048
          int l     = m & 2047;
          bf16_t bv = (bf16_t)v;
          size_t bh = (size_t)(b * C_NH + head);
          if (which == 0)      Qb[(bh * C_L + l) * C_HS + d] = bv;
          else if (which == 1) Kb[(bh * C_L + l) * C_HS + d] = bv;
          else                 Vt[(bh * C_HS + d) * C_L + l] = bv;
        }
      }
}

// ---------------------------------------------------------------------------
// Flash attention: grid (L/128, NH, B); 8 waves x 16 query rows.
// S = (Q K^T)/8 + alibi + gamma*adj ; online softmax ; O = P V.
// Double-buffered K/V tiles via async DMA. Output bf16 [B*L, H].
// ---------------------------------------------------------------------------
__global__ __launch_bounds__(256) void k_flash_attn(
    const bf16_t* __restrict__ Qb, const bf16_t* __restrict__ Kb,
    const bf16_t* __restrict__ Vt, const float* __restrict__ adj,
    const float* __restrict__ gamma, bf16_t* __restrict__ Ob) {
  const int bm   = blockIdx.x;
  const int h    = blockIdx.y;
  const int b    = blockIdx.z;
  const int tid  = threadIdx.x;
  const int lane = tid & 31;
  const int wave = tid >> 5;
  const int lm   = lane & 15;
  const int hi   = lane >> 4;
  const int kh   = hi * 8;

  __shared__ __align__(16) bf16_t Ksm[2][64][64];    // [key][hs]
  __shared__ __align__(16) bf16_t Vsm[2][64][64];    // [hs][key]
  __shared__ __align__(16) bf16_t Psm[8][16][64];    // per-wave P staging

  const size_t  bh   = (size_t)(b * C_NH + h);
  const bf16_t* Qp   = Qb + bh * C_L * C_HS;
  const bf16_t* Kp   = Kb + bh * C_L * C_HS;
  const bf16_t* Vp   = Vt + bh * C_HS * C_L;
  const float*  adjp = adj + (size_t)b * C_L * C_L;

  const int   mrow0 = bm * 128 + wave * 16;
  const float slope = (h < 8) ? exp2f(-(float)(h + 1)) : 0.0f;
  const float g     = gamma[h];
  const float scale = 0.125f;   // 1/sqrt(64)

  auto load_tile = [&](int j0, int buf) {
#pragma unroll
    for (int i = 0; i < 2; ++i) {           // 512 16B chunks per K/V tile pair
      int c = tid + 256 * i;
      int r = c >> 3, cc = (c & 7) * 8;
      cp16(&Kp[((size_t)(j0 + r)) * C_HS + cc], &Ksm[buf][r][cc]);
      cp16(&Vp[(size_t)r * C_L + j0 + cc], &Vsm[buf][r][cc]);
    }
  };

  // Q fragments for this wave's 16 rows (resident all loop)
  v16bf qf[2];
  {
    int row = mrow0 + lm;
#pragma unroll
    for (int ks = 0; ks < 2; ++ks) {
      const bf16_t* p = &Qp[(size_t)row * C_HS + ks * 32 + kh];
      qf[ks] = ld_frag(p, p + 16);
    }
  }

  float mr[8], lr[8];
#pragma unroll
  for (int r = 0; r < 8; ++r) { mr[r] = -1e30f; lr[r] = 0.0f; }
  v8f o[4];
#pragma unroll
  for (int ns = 0; ns < 4; ++ns) o[ns] = (v8f)0.0f;

  load_tile(0, 0);
  wait_async_all();
  __syncthreads();

  int cur = 0;
  for (int j0 = 0; j0 < C_L; j0 += 64) {
    if (j0 + 64 < C_L) load_tile(j0 + 64, cur ^ 1);   // DMA overlaps compute

    // S = Q K^T  (wave: 16 rows x 64 keys)
    v8f s[4];
#pragma unroll
    for (int ns = 0; ns < 4; ++ns) s[ns] = (v8f)0.0f;
#pragma unroll
    for (int ns = 0; ns < 4; ++ns)
#pragma unroll
      for (int ks = 0; ks < 2; ++ks) {
        const bf16_t* p = &Ksm[cur][ns * 16 + lm][ks * 32 + kh];
        s[ns] = wmma_bf16(qf[ks], ld_frag(p, p + 16), s[ns]);
      }

    // scale + alibi + gamma*adj
    float sv[4][8];
#pragma unroll
    for (int ns = 0; ns < 4; ++ns)
#pragma unroll
      for (int r = 0; r < 8; ++r) {
        int m = mrow0 + r + 8 * hi;
        int n = j0 + ns * 16 + lm;
        sv[ns][r] = s[ns][r] * scale - slope * fabsf((float)(m - n))
                    + g * adjp[(size_t)m * C_L + n];
      }

    // row-wise max (rows live in VGPR index r, cols in lanes of each half)
    float tm[8];
#pragma unroll
    for (int r = 0; r < 8; ++r) {
      float v = sv[0][r];
      v = fmaxf(v, sv[1][r]); v = fmaxf(v, sv[2][r]); v = fmaxf(v, sv[3][r]);
      tm[r] = v;
    }
#pragma unroll
    for (int off = 1; off < 16; off <<= 1)
#pragma unroll
      for (int r = 0; r < 8; ++r) tm[r] = fmaxf(tm[r], __shfl_xor(tm[r], off, 32));

    float alpha[8], rs[8];
#pragma unroll
    for (int r = 0; r < 8; ++r) {
      float mn = fmaxf(mr[r], tm[r]);
      alpha[r] = __expf(mr[r] - mn);
      mr[r] = mn;
      rs[r] = 0.0f;
    }
#pragma unroll
    for (int ns = 0; ns < 4; ++ns)
#pragma unroll
      for (int r = 0; r < 8; ++r) {
        float p = __expf(sv[ns][r] - mr[r]);
        sv[ns][r] = p;
        rs[r] += p;
      }
#pragma unroll
    for (int off = 1; off < 16; off <<= 1)
#pragma unroll
      for (int r = 0; r < 8; ++r) rs[r] += __shfl_xor(rs[r], off, 32);
#pragma unroll
    for (int r = 0; r < 8; ++r) lr[r] = lr[r] * alpha[r] + rs[r];
#pragma unroll
    for (int ns = 0; ns < 4; ++ns)
#pragma unroll
      for (int r = 0; r < 8; ++r) o[ns][r] *= alpha[r];

    // re-layout P: C-fragment -> A-fragment via per-wave LDS staging
#pragma unroll
    for (int ns = 0; ns < 4; ++ns)
#pragma unroll
      for (int r = 0; r < 8; ++r)
        Psm[wave][r + 8 * hi][ns * 16 + lm] = (bf16_t)sv[ns][r];
    asm volatile("s_wait_dscnt 0" ::: "memory");   // wave-local RAW on LDS

    // O += P V
    v16bf pf[2];
#pragma unroll
    for (int ks = 0; ks < 2; ++ks) {
      const bf16_t* p = &Psm[wave][lm][ks * 32 + kh];
      pf[ks] = ld_frag(p, p + 16);
    }
#pragma unroll
    for (int ns = 0; ns < 4; ++ns)
#pragma unroll
      for (int ks = 0; ks < 2; ++ks) {
        const bf16_t* p = &Vsm[cur][ns * 16 + lm][ks * 32 + kh];
        o[ns] = wmma_bf16(pf[ks], ld_frag(p, p + 16), o[ns]);
      }

    wait_async_all();
    __syncthreads();
    cur ^= 1;
  }

  // finalize: O /= l ; store bf16 [B*L, H]
  float inv[8];
#pragma unroll
  for (int r = 0; r < 8; ++r) inv[r] = 1.0f / lr[r];
#pragma unroll
  for (int ns = 0; ns < 4; ++ns)
#pragma unroll
    for (int r = 0; r < 8; ++r) {
      int m = mrow0 + r + 8 * hi;
      int d = ns * 16 + lm;
      Ob[((size_t)(b * C_L + m)) * C_H + h * C_HS + d] = (bf16_t)(o[ns][r] * inv[r]);
    }
}

// ---------------------------------------------------------------------------
// Launch
// ---------------------------------------------------------------------------
extern "C" void kernel_launch(void* const* d_in, const int* in_sizes, int n_in,
                              void* d_out, int out_size, void* d_ws, size_t ws_size,
                              hipStream_t stream) {
  const float* x        = (const float*)d_in[0];   // [B,L,H]
  const float* adj      = (const float*)d_in[1];   // [B,1,L,L]
  const float* weights  = (const float*)d_in[2];   // [H,3H]
  const float* in_bias  = (const float*)d_in[3];   // [3H]
  const float* out_w    = (const float*)d_in[4];   // [H,H]
  const float* out_bias = (const float*)d_in[5];   // [H]
  const float* gamma    = (const float*)d_in[6];   // [NH]
  float* out = (float*)d_out;

  const int M  = C_B * C_L;        // 4096
  const int N1 = 3 * C_H;          // 3072
  const int K  = C_H;              // 1024

  char* ws = (char*)d_ws;
  bf16_t* xb   = (bf16_t*)ws;  ws += (size_t)M * C_H * 2;
  bf16_t* wT   = (bf16_t*)ws;  ws += (size_t)N1 * C_H * 2;
  bf16_t* owT  = (bf16_t*)ws;  ws += (size_t)C_H * C_H * 2;
  bf16_t* Qb   = (bf16_t*)ws;  ws += (size_t)C_B * C_NH * C_L * C_HS * 2;
  bf16_t* Kb   = (bf16_t*)ws;  ws += (size_t)C_B * C_NH * C_L * C_HS * 2;
  bf16_t* Vt   = (bf16_t*)ws;  ws += (size_t)C_B * C_NH * C_L * C_HS * 2;
  bf16_t* attb = (bf16_t*)ws;  ws += (size_t)M * C_H * 2;

  // stage 1: conversions
  {
    int n = M * C_H;
    k_cvt_bf16<<<(n + 255) / 256, 256, 0, stream>>>(x, xb, n);
  }
  k_transpose_bf16<<<(C_H * N1 + 255) / 256, 256, 0, stream>>>(weights, wT, C_H, N1);
  k_transpose_bf16<<<(C_H * C_H + 255) / 256, 256, 0, stream>>>(out_w, owT, C_H, C_H);

  // stage 2: QKV projection (scatter epilogue)
  k_gemm_bf16<<<dim3(N1 / TN, M / TM), 256, 0, stream>>>(
      xb, wT, in_bias, M, N1, K, /*mode=*/0, Qb, Kb, Vt, nullptr);

  // stage 3: flash attention
  k_flash_attn<<<dim3(C_L / 128, C_NH, C_B), 256, 0, stream>>>(
      Qb, Kb, Vt, adj, gamma, attb);

  // stage 4: output projection + bias -> fp32
  k_gemm_bf16<<<dim3(C_H / TN, M / TM), 256, 0, stream>>>(
      attb, owT, out_bias, M, C_H, K, /*mode=*/1, nullptr, nullptr, nullptr, out);
}
